// LSTM_6150393167884
// MI455X (gfx1250) — matrix operand order repaired and checked by reading, not compiled, not run on hardware
//
#include <hip/hip_runtime.h>

#define BDIM 64
#define TDIM 512
#define IDIM 512
#define HDIM 1024
#define NCOL 4096          // 4 gates * H
#define KTOT 1536          // I + H
#define KT_X 16            // K-tiles (of 32) covering the x region: 512/32
#define KT_ALL 48          // 1536/32
#define NTILES 256         // 4096/16

typedef __bf16 bf16_t;
typedef bf16_t v16bf __attribute__((ext_vector_type(16)));
typedef bf16_t v8bf  __attribute__((ext_vector_type(8)));
typedef float  v8f   __attribute__((ext_vector_type(8)));

// ---------------- init: h(bf16)=0, c(f32)=0 ----------------
__global__ void lstm_init_state(bf16_t* __restrict__ hbf, float* __restrict__ c) {
    int i = blockIdx.x * blockDim.x + threadIdx.x;
    if (i < BDIM * HDIM) { hbf[i] = (bf16_t)0.0f; c[i] = 0.0f; }
}

// ---------------- convert x f32 -> bf16 ----------------
__global__ void lstm_cvt_x(const float* __restrict__ x, bf16_t* __restrict__ xb, int n) {
    int i = blockIdx.x * blockDim.x + threadIdx.x;
    int stride = gridDim.x * blockDim.x;
    for (; i < n; i += stride) xb[i] = (bf16_t)x[i];
}

// ---------------- pack Z = [W^T ; U^T] into wave32 WMMA B-fragment layout ----------------
// Packed element (tile = kt*NTILES+nt, lane, e) holds Z[k][n] with
//   k = kt*32 + (e>>3)*16 + (lane>>4)*8 + (e&7),  n = nt*16 + (lane&15)   (16-bit B layout, wave32)
__global__ void lstm_pack_weights(const float* __restrict__ Wi, const float* __restrict__ Wf,
                                  const float* __restrict__ Wc, const float* __restrict__ Wo,
                                  const float* __restrict__ Ui, const float* __restrict__ Uf,
                                  const float* __restrict__ Uc, const float* __restrict__ Uo,
                                  bf16_t* __restrict__ Zp) {
    int idx = blockIdx.x * blockDim.x + threadIdx.x;
    if (idx >= KT_ALL * NTILES * 512) return;
    int tile = idx >> 9;
    int r    = idx & 511;
    int lane = r >> 4;
    int e    = r & 15;
    int kt   = tile >> 8;           // /NTILES
    int nt   = tile & (NTILES - 1);
    int k = kt * 32 + ((e >> 3) << 4) + ((lane >> 4) << 3) + (e & 7);
    int n = nt * 16 + (lane & 15);
    int g = n >> 10;                // gate: 0=i 1=f 2=g(c) 3=o
    int h = n & (HDIM - 1);
    float v;
    if (k < IDIM) {
        const float* W = (g == 0) ? Wi : (g == 1) ? Wf : (g == 2) ? Wc : Wo;
        v = W[h * IDIM + k];
    } else {
        const float* U = (g == 0) ? Ui : (g == 1) ? Uf : (g == 2) ? Uc : Uo;
        v = U[h * HDIM + (k - IDIM)];
    }
    Zp[idx] = (bf16_t)v;
}

// A fragment loader (16-bit A 16x32 layout): two contiguous 16B chunks per lane,
// covering K = kk..kk+7 and K = kk+16..kk+23 for this lane's row.
__device__ __forceinline__ v16bf lstm_load_a(const bf16_t* __restrict__ p) {
    union { v16bf v; v8bf h2[2]; } u;
    u.h2[0] = *(const v8bf*)p;
    u.h2[1] = *(const v8bf*)(p + 16);
    return u.v;
}

// ---------------- per-timestep fused GEMM: pre = [x_t | h_prev] @ Z ----------------
// Grid: 64 blocks * 128 threads (4 waves). Each wave computes the FULL M=64 strip
// of one 16-column N tile with 4 independent accumulators, so each B fragment
// (L2-resident packed weights) is reused by 4 WMMAs and consecutive WMMAs have
// no accumulator RAW hazard.
__global__ __launch_bounds__(128)
void lstm_step_gemm(const bf16_t* __restrict__ xb, const bf16_t* __restrict__ hbf,
                    const bf16_t* __restrict__ Zp, float* __restrict__ pre, int t) {
    const int lane  = threadIdx.x & 31;
    const int wave  = threadIdx.x >> 5;
    const int ntile = blockIdx.x * 4 + wave;   // 0..255
    const int n0 = ntile * 16;
    const int hi = lane >> 4;
    const int ml = lane & 15;

    v8f acc0 = {}, acc1 = {}, acc2 = {}, acc3 = {};

    // ---- x region: K tiles 0..15 ----
    {
        const bf16_t* r0 = xb + ((size_t)(ml +  0) * TDIM + (size_t)t) * IDIM;
        const bf16_t* r1 = xb + ((size_t)(ml + 16) * TDIM + (size_t)t) * IDIM;
        const bf16_t* r2 = xb + ((size_t)(ml + 32) * TDIM + (size_t)t) * IDIM;
        const bf16_t* r3 = xb + ((size_t)(ml + 48) * TDIM + (size_t)t) * IDIM;
        for (int kt = 0; kt < KT_X; ++kt) {
            const bf16_t* bp = Zp + (((size_t)kt * NTILES + (size_t)ntile) << 9) + ((size_t)lane << 4);
            v16bf bfrag = *(const v16bf*)bp;
            __builtin_prefetch(bp + ((size_t)NTILES << 9), 0, 3);
            int kk = kt * 32 + hi * 8;
            v16bf a0 = lstm_load_a(r0 + kk);
            v16bf a1 = lstm_load_a(r1 + kk);
            v16bf a2 = lstm_load_a(r2 + kk);
            v16bf a3 = lstm_load_a(r3 + kk);
            acc0 = __builtin_amdgcn_wmma_f32_16x16x32_bf16(false, a0, false, bfrag, (short)0, acc0, false, false);
            acc1 = __builtin_amdgcn_wmma_f32_16x16x32_bf16(false, a1, false, bfrag, (short)0, acc1, false, false);
            acc2 = __builtin_amdgcn_wmma_f32_16x16x32_bf16(false, a2, false, bfrag, (short)0, acc2, false, false);
            acc3 = __builtin_amdgcn_wmma_f32_16x16x32_bf16(false, a3, false, bfrag, (short)0, acc3, false, false);
        }
    }

    // ---- h region: K tiles 16..47 ----
    {
        const bf16_t* r0 = hbf + (size_t)(ml +  0) * HDIM;
        const bf16_t* r1 = hbf + (size_t)(ml + 16) * HDIM;
        const bf16_t* r2 = hbf + (size_t)(ml + 32) * HDIM;
        const bf16_t* r3 = hbf + (size_t)(ml + 48) * HDIM;
        for (int kt = KT_X; kt < KT_ALL; ++kt) {
            const bf16_t* bp = Zp + (((size_t)kt * NTILES + (size_t)ntile) << 9) + ((size_t)lane << 4);
            v16bf bfrag = *(const v16bf*)bp;
            if (kt + 1 < KT_ALL) __builtin_prefetch(bp + ((size_t)NTILES << 9), 0, 3);
            int kk = kt * 32 + hi * 8 - IDIM;
            v16bf a0 = lstm_load_a(r0 + kk);
            v16bf a1 = lstm_load_a(r1 + kk);
            v16bf a2 = lstm_load_a(r2 + kk);
            v16bf a3 = lstm_load_a(r3 + kk);
            acc0 = __builtin_amdgcn_wmma_f32_16x16x32_bf16(false, a0, false, bfrag, (short)0, acc0, false, false);
            acc1 = __builtin_amdgcn_wmma_f32_16x16x32_bf16(false, a1, false, bfrag, (short)0, acc1, false, false);
            acc2 = __builtin_amdgcn_wmma_f32_16x16x32_bf16(false, a2, false, bfrag, (short)0, acc2, false, false);
            acc3 = __builtin_amdgcn_wmma_f32_16x16x32_bf16(false, a3, false, bfrag, (short)0, acc3, false, false);
        }
    }

    // D layout per tile: VGPR r -> row mbase + hi*8 + r, col n0 + (lane&15)
    {
        float* o0 = pre + (size_t)( 0 + hi * 8) * NCOL + (size_t)n0 + ml;
        float* o1 = pre + (size_t)(16 + hi * 8) * NCOL + (size_t)n0 + ml;
        float* o2 = pre + (size_t)(32 + hi * 8) * NCOL + (size_t)n0 + ml;
        float* o3 = pre + (size_t)(48 + hi * 8) * NCOL + (size_t)n0 + ml;
        #pragma unroll
        for (int r = 0; r < 8; ++r) {
            o0[(size_t)r * NCOL] = acc0[r];
            o1[(size_t)r * NCOL] = acc1[r];
            o2[(size_t)r * NCOL] = acc2[r];
            o3[(size_t)r * NCOL] = acc3[r];
        }
    }
}

// ---------------- per-timestep pointwise: gates, cell update, outputs ----------------
__global__ void lstm_step_pointwise(const float* __restrict__ pre,
                                    const float* __restrict__ bi, const float* __restrict__ bfv,
                                    const float* __restrict__ bc, const float* __restrict__ bo,
                                    float* __restrict__ c, bf16_t* __restrict__ hbf,
                                    float* __restrict__ out, int t, int last) {
    int idx = blockIdx.x * blockDim.x + threadIdx.x;
    if (idx >= BDIM * HDIM) return;
    int b = idx >> 10;
    int h = idx & (HDIM - 1);
    const float* p = pre + (size_t)b * NCOL;
    float xi = p[h]            + bi[h];
    float xf = p[HDIM + h]     + bfv[h];
    float xg = p[2 * HDIM + h] + bc[h];
    float xo = p[3 * HDIM + h] + bo[h];
    float ig = 1.0f / (1.0f + __expf(-xi));
    float fg = 1.0f / (1.0f + __expf(-xf));
    float gg = tanhf(xg);
    float og = 1.0f / (1.0f + __expf(-xo));
    float cn = fg * c[idx] + ig * gg;
    float hn = og * tanhf(cn);
    c[idx]   = cn;
    hbf[idx] = (bf16_t)hn;
    out[((size_t)b * TDIM + (size_t)t) * HDIM + h] = hn;
    if (last) {
        size_t tail = (size_t)BDIM * TDIM * HDIM;
        out[tail + idx] = hn;                       // h_n
        out[tail + (size_t)BDIM * HDIM + idx] = cn; // c_n
    }
}

extern "C" void kernel_launch(void* const* d_in, const int* in_sizes, int n_in,
                              void* d_out, int out_size, void* d_ws, size_t ws_size,
                              hipStream_t stream) {
    const float* x   = (const float*)d_in[0];
    const float* Wi  = (const float*)d_in[1];
    const float* Wf  = (const float*)d_in[2];
    const float* Wc  = (const float*)d_in[3];
    const float* Wo  = (const float*)d_in[4];
    const float* Ui  = (const float*)d_in[5];
    const float* Uf  = (const float*)d_in[6];
    const float* Uc  = (const float*)d_in[7];
    const float* Uo  = (const float*)d_in[8];
    const float* bi  = (const float*)d_in[9];
    const float* bfv = (const float*)d_in[10];
    const float* bc  = (const float*)d_in[11];
    const float* bo  = (const float*)d_in[12];
    float* out = (float*)d_out;

    // carve workspace (256B aligned)
    char* ws = (char*)d_ws;
    size_t off = 0;
    auto carve = [&](size_t bytes) -> void* {
        void* p = ws + off;
        off += (bytes + 255) & ~(size_t)255;
        return p;
    };
    bf16_t* xb  = (bf16_t*)carve((size_t)BDIM * TDIM * IDIM * sizeof(bf16_t)); // 32 MB
    bf16_t* Zp  = (bf16_t*)carve((size_t)KTOT * NCOL * sizeof(bf16_t));        // 12 MB
    bf16_t* hbf = (bf16_t*)carve((size_t)BDIM * HDIM * sizeof(bf16_t));
    float*  c   = (float*) carve((size_t)BDIM * HDIM * sizeof(float));
    float*  pre = (float*) carve((size_t)BDIM * NCOL * sizeof(float));         // 1 MB

    lstm_init_state<<<(BDIM * HDIM + 255) / 256, 256, 0, stream>>>(hbf, c);
    lstm_cvt_x<<<4096, 256, 0, stream>>>(x, xb, BDIM * TDIM * IDIM);
    lstm_pack_weights<<<((size_t)KTOT * NCOL + 255) / 256, 256, 0, stream>>>(
        Wi, Wf, Wc, Wo, Ui, Uf, Uc, Uo, Zp);

    for (int t = 0; t < TDIM; ++t) {
        lstm_step_gemm<<<64, 128, 0, stream>>>(xb, hbf, Zp, pre, t);
        lstm_step_pointwise<<<(BDIM * HDIM + 255) / 256, 256, 0, stream>>>(
            pre, bi, bfv, bc, bo, c, hbf, out, t, (t == TDIM - 1) ? 1 : 0);
    }
}
